// ODEFunc_88055419503310
// MI455X (gfx1250) — matrix operand order repaired and checked by reading, not compiled, not run on hardware
//
#include <hip/hip_runtime.h>
#include <hip/hip_bf16.h>

typedef __attribute__((ext_vector_type(16))) _Float16 v16h;
typedef __attribute__((ext_vector_type(8)))  _Float16 v8h;
typedef __attribute__((ext_vector_type(8)))  float    v8f;

#define B_SZ   4096
#define N_PTS  256
#define ROWLEN 13376   // padded theta row: 240 | 6480 | 6480 | 176
#define T0_OFF 0
#define T1_OFF 240
#define T2_OFF 6720
#define T5_OFF 13200

// fast silu: one v_exp + one v_rcp instead of the IEEE division chain
__device__ __forceinline__ float silu_f(float x) {
    return x * __builtin_amdgcn_rcpf(1.0f + __expf(-x));
}

__device__ __forceinline__ v16h pack16(v8h lo, v8h hi) {
    return __builtin_shufflevector(lo, hi, 0,1,2,3,4,5,6,7,8,9,10,11,12,13,14,15);
}

// ---------------------------------------------------------------------------
// Kernel 1: dw_dt, zeros, encoder w(16)->e1(50)->we(200), hidden s1/s2 (50),
// emit K-padded f16 activations S[b][256] = [s0|s1|s2|s5] (4 x 64-wide segs).
// One 64-thread block per batch; weights stay L2-resident.
// ---------------------------------------------------------------------------
__global__ void __launch_bounds__(64)
prep_kernel(const float* __restrict__ w, const float* __restrict__ a,
            const float* __restrict__ encW1, const float* __restrict__ encb1,
            const float* __restrict__ encW2, const float* __restrict__ encb2,
            const float* __restrict__ e1W1, const float* __restrict__ e1b1,
            const float* __restrict__ e2W1, const float* __restrict__ e2b1,
            _Float16* __restrict__ S, float* __restrict__ out_dw, float* __restrict__ out_z)
{
    __shared__ float wss[16], ass[16], e1s[50], wcs[200];
    const int b = blockIdx.x;
    const int t = threadIdx.x;
    if (t < 16) {
        wss[t] = w[b * 16 + t];
        ass[t] = a[b * 16 + t];
        out_z[b * 16 + t] = 0.0f;
    }
    __syncthreads();
    if (t < 8) {  // dw_dt from complex-style pairing
        float re = ass[t], im = ass[8 + t];
        float we_ = wss[2 * t], wo = wss[2 * t + 1];
        out_dw[b * 16 + 2 * t]     = we_ * re + wo * im;
        out_dw[b * 16 + 2 * t + 1] = wo * re - we_ * im;
    }
    if (t < 50) {
        float s = encb1[t];
#pragma unroll
        for (int k = 0; k < 16; ++k) s += wss[k] * encW1[k * 50 + t];
        e1s[t] = silu_f(s);
    }
    __syncthreads();
    for (int j = t; j < 200; j += 64) {
        float s = encb2[j];
        for (int k = 0; k < 50; ++k) s += e1s[k] * encW2[k * 200 + j];
        wcs[j] = s;
    }
    __syncthreads();
    _Float16* Sr = S + (size_t)b * 256;
    // seg0 = we[0:50], seg3 = we[150:200] (raw); seg1/seg2 = silu(hidden)
    Sr[t]       = (t < 50) ? (_Float16)wcs[t]       : (_Float16)0.0f;
    Sr[192 + t] = (t < 50) ? (_Float16)wcs[150 + t] : (_Float16)0.0f;
    float v1 = 0.0f, v2 = 0.0f;
    if (t < 50) {
        float s1 = e1b1[t], s2 = e2b1[t];
        for (int k = 0; k < 50; ++k) {
            s1 += wcs[50 + k]  * e1W1[k * 50 + t];
            s2 += wcs[100 + k] * e2W1[k * 50 + t];
        }
        v1 = silu_f(s1); v2 = silu_f(s2);
    }
    Sr[64 + t]  = (_Float16)v1;
    Sr[128 + t] = (_Float16)v2;
}

// ---------------------------------------------------------------------------
// Kernel 2: head weights (50 x N, f32, k-major) -> f16 WT[13376][64],
// n-major with K padded 50->64 (B-fragment friendly: column-contiguous K).
// ---------------------------------------------------------------------------
__global__ void __launch_bounds__(256)
convert_kernel(const float* __restrict__ e0W, const float* __restrict__ e1W2,
               const float* __restrict__ e2W2, const float* __restrict__ e5W,
               _Float16* __restrict__ WT)
{
    int idx = blockIdx.x * blockDim.x + threadIdx.x;
    if (idx >= 13376 * 64) return;
    int ng = idx >> 6;
    int k  = idx & 63;
    float v = 0.0f;
    if (k < 50) {
        if (ng < 240)        v = e0W[k * 240 + ng];
        else if (ng < 6720)  v = e1W2[k * 6480 + (ng - 240)];
        else if (ng < 13200) v = e2W2[k * 6480 + (ng - 6720)];
        else { int n = ng - 13200; if (n < 162) v = e5W[k * 162 + n]; }
    }
    WT[idx] = (_Float16)v;
}

// ---------------------------------------------------------------------------
// Kernel 3: theta = act(S @ WT + bias), M=4096, K=64, N per head.
// One 16x16 tile per wave; bias preloaded into the accumulator; f16 out.
// ---------------------------------------------------------------------------
__global__ void __launch_bounds__(256)
theta_gemm_kernel(const _Float16* __restrict__ S, const _Float16* __restrict__ WT,
                  const float* __restrict__ bias, _Float16* __restrict__ theta,
                  int segA, int wtRow0, int Nvalid, int Npad, int thetaOff, int applySilu)
{
    const int lane = threadIdx.x & 31;
    const int wv   = threadIdx.x >> 5;
    const int ntile = blockIdx.y * 8 + wv;
    if (ntile * 16 >= Npad) return;           // wave-uniform -> EXEC stays all-1
    const int mtile = blockIdx.x;
    const int nl = lane & 15;
    const int n  = ntile * 16 + nl;
    const float bval = (n < Nvalid) ? bias[n] : 0.0f;
    v8f c;
#pragma unroll
    for (int r = 0; r < 8; ++r) c[r] = bval;
    const _Float16* Abase = S + ((size_t)mtile * 16 + nl) * 256 + segA;
    const _Float16* Wbase = WT + (size_t)(wtRow0 + n) * 64 + ((lane >= 16) ? 16 : 0);
    const int kbA = (lane >= 16) ? 16 : 0;
#pragma unroll
    for (int s = 0; s < 2; ++s) {
        const _Float16* pa = Abase + s * 32 + kbA;
        v16h av = pack16(*(const v8h*)pa, *(const v8h*)(pa + 8));
        v16h bv = *(const v16h*)(Wbase + s * 32);
        c = __builtin_amdgcn_wmma_f32_16x16x32_f16(false, av, false, bv, (short)0, c, false, false);
    }
    if (applySilu) {
#pragma unroll
        for (int r = 0; r < 8; ++r) c[r] = silu_f(c[r]);
    }
    const int mbase = mtile * 16 + ((lane >= 16) ? 8 : 0);
    const size_t col = (size_t)thetaOff + ntile * 16 + nl;
#pragma unroll
    for (int r = 0; r < 8; ++r)
        theta[(size_t)(mbase + r) * ROWLEN + col] = (_Float16)c[r];
}

// ---------------------------------------------------------------------------
// Per-batch MLP: one block = one batch, 8 waves. Wave owns 32 rows of h;
// layers 1&2 are WMMA with accumulators kept in registers (in-place LDS h).
// A fragments hoisted out of the n-tile loop (depend only on m-tile, k-step).
// ---------------------------------------------------------------------------
__device__ __forceinline__ void
layer_wmma(_Float16* hA, const _Float16* WTs, const _Float16* biasL, int wv, int lane)
{
    v8f acc[10];
    const int nl = lane & 15;
    const int kbA = (lane >= 16) ? 16 : 0;
    const int kbB = (lane >= 16) ? 16 : 0;
    const int mrow0 = wv * 32;
#pragma unroll
    for (int mt = 0; mt < 2; ++mt) {
        const _Float16* arow = hA + (mrow0 + mt * 16 + nl) * 96;
        v16h av[3];
#pragma unroll
        for (int s = 0; s < 3; ++s) {
            const _Float16* pa = arow + s * 32 + kbA;
            av[s] = pack16(*(const v8h*)pa, *(const v8h*)(pa + 8));
        }
#pragma unroll
        for (int nt = 0; nt < 5; ++nt) {
            const int n = nt * 16 + nl;
            const float bv = (float)biasL[n];
            v8f c;
#pragma unroll
            for (int r = 0; r < 8; ++r) c[r] = bv;
            const _Float16* brow = WTs + n * 96 + kbB;
#pragma unroll
            for (int s = 0; s < 3; ++s) {
                v16h bf = *(const v16h*)(brow + s * 32);
                c = __builtin_amdgcn_wmma_f32_16x16x32_f16(false, av[s], false, bf, (short)0, c, false, false);
            }
#pragma unroll
            for (int r = 0; r < 8; ++r) c[r] = fmaxf(c[r], 0.0f);  // relu
            acc[mt * 5 + nt] = c;
        }
    }
    // write back (reads of this wave's rows all completed above; rows are wave-private)
#pragma unroll
    for (int mt = 0; mt < 2; ++mt) {
        const int mb = mrow0 + mt * 16 + ((lane >= 16) ? 8 : 0);
#pragma unroll
        for (int nt = 0; nt < 5; ++nt) {
            v8f c = acc[mt * 5 + nt];
#pragma unroll
            for (int r = 0; r < 8; ++r)
                hA[(mb + r) * 96 + nt * 16 + nl] = (_Float16)c[r];
        }
    }
}

__global__ void __launch_bounds__(256)
mlp_kernel(const float* __restrict__ y, const _Float16* __restrict__ theta,
           float* __restrict__ dy)
{
    __shared__ _Float16 hA[256 * 96];     // 48 KB, K padded 80->96
    __shared__ _Float16 WTs[80 * 96];     // 15 KB, W1 then W2, n-major
    __shared__ _Float16 w03[240];         // W0/b0, later W3/b3
    __shared__ _Float16 biasL[80];        // b1 then b2
    const int b = blockIdx.x;
    const int t = threadIdx.x;
    const int lane = t & 31;
    const int wv = t >> 5;
    const _Float16* th = theta + (size_t)b * ROWLEN;

    // warm L2/WGP$ for the W2 block while we work on layers 0-1
    // (256 lanes x 128B lines cover the 12.8KB theta2 region)
    __builtin_prefetch(th + T2_OFF + t * 64, 0, 1);

    // stage W1 (transposed, zero-padded), b1, W0/b0
    for (int i = t; i < 240; i += 256) w03[i] = th[T0_OFF + i];
    for (int i = t; i < 80 * 96; i += 256) {
        int n = i / 96, k = i % 96;
        WTs[i] = (k < 80) ? th[T1_OFF + k * 80 + n] : (_Float16)0.0f;
    }
    if (t < 80) biasL[t] = th[T1_OFF + 6400 + t];
    __syncthreads();

    // layer 0 (K=2) scalar -> hA f16
    {
        const size_t yo = (((size_t)b * 256) + t) * 2;
        const float y0 = y[yo], y1 = y[yo + 1];
        _Float16* hr = hA + t * 96;
#pragma unroll
        for (int j = 0; j < 80; ++j) {
            float h = y0 * (float)w03[j] + y1 * (float)w03[80 + j] + (float)w03[160 + j];
            hr[j] = (_Float16)fmaxf(h, 0.0f);
        }
#pragma unroll
        for (int j = 80; j < 96; ++j) hr[j] = (_Float16)0.0f;
    }
    __syncthreads();

    layer_wmma(hA, WTs, biasL, wv, lane);      // layer 1
    __syncthreads();

    // stage W2, b2, W3/b3
    for (int i = t; i < 80 * 96; i += 256) {
        int n = i / 96, k = i % 96;
        WTs[i] = (k < 80) ? th[T2_OFF + k * 80 + n] : (_Float16)0.0f;
    }
    if (t < 80) biasL[t] = th[T2_OFF + 6400 + t];
    for (int i = t; i < 162; i += 256) w03[i] = th[T5_OFF + i];
    __syncthreads();

    layer_wmma(hA, WTs, biasL, wv, lane);      // layer 2
    __syncthreads();

    // layer 3 (N=2) scalar
    {
        float a0 = (float)w03[160], a1 = (float)w03[161];
        const _Float16* hr = hA + t * 96;
#pragma unroll 8
        for (int h = 0; h < 80; ++h) {
            float hv = (float)hr[h];
            a0 += hv * (float)w03[h * 2 + 0];
            a1 += hv * (float)w03[h * 2 + 1];
        }
        const size_t o = (((size_t)b * 256) + t) * 2;
        dy[o] = a0; dy[o + 1] = a1;
    }
}

// ---------------------------------------------------------------------------
extern "C" void kernel_launch(void* const* d_in, const int* in_sizes, int n_in,
                              void* d_out, int out_size, void* d_ws, size_t ws_size,
                              hipStream_t stream)
{
    (void)in_sizes; (void)n_in; (void)out_size; (void)ws_size;
    const float* y     = (const float*)d_in[1];
    const float* w     = (const float*)d_in[2];
    const float* a     = (const float*)d_in[3];
    const float* encW1 = (const float*)d_in[4];
    const float* encb1 = (const float*)d_in[5];
    const float* encW2 = (const float*)d_in[6];
    const float* encb2 = (const float*)d_in[7];
    const float* e0W   = (const float*)d_in[8];
    const float* e0b   = (const float*)d_in[9];
    const float* e1W1  = (const float*)d_in[10];
    const float* e1b1  = (const float*)d_in[11];
    const float* e1W2  = (const float*)d_in[12];
    const float* e1b2  = (const float*)d_in[13];
    const float* e2W1  = (const float*)d_in[14];
    const float* e2b1  = (const float*)d_in[15];
    const float* e2W2  = (const float*)d_in[16];
    const float* e2b2  = (const float*)d_in[17];
    const float* e5W   = (const float*)d_in[18];
    const float* e5b   = (const float*)d_in[19];

    float* out    = (float*)d_out;
    float* out_dy = out;
    float* out_dw = out + (size_t)B_SZ * N_PTS * 2;
    float* out_z  = out_dw + (size_t)B_SZ * 16;

    char* ws = (char*)d_ws;
    _Float16* S     = (_Float16*)(ws);                       // 2,097,152 B
    _Float16* WT    = (_Float16*)(ws + 2097152);             // 1,712,128 B
    _Float16* theta = (_Float16*)(ws + 2097152 + 1712128);   // 109,576,192 B

    prep_kernel<<<B_SZ, 64, 0, stream>>>(w, a, encW1, encb1, encW2, encb2,
                                         e1W1, e1b1, e2W1, e2b1, S, out_dw, out_z);
    convert_kernel<<<(13376 * 64 + 255) / 256, 256, 0, stream>>>(e0W, e1W2, e2W2, e5W, WT);
    theta_gemm_kernel<<<dim3(256, 2),  256, 0, stream>>>(S, WT, e0b,  theta,   0,     0,  240,  240, T0_OFF, 1);
    theta_gemm_kernel<<<dim3(256, 51), 256, 0, stream>>>(S, WT, e1b2, theta,  64,   240, 6480, 6480, T1_OFF, 0);
    theta_gemm_kernel<<<dim3(256, 51), 256, 0, stream>>>(S, WT, e2b2, theta, 128,  6720, 6480, 6480, T2_OFF, 0);
    theta_gemm_kernel<<<dim3(256, 2),  256, 0, stream>>>(S, WT, e5b,  theta, 192, 13200,  162,  176, T5_OFF, 1);
    mlp_kernel<<<B_SZ, 256, 0, stream>>>(y, theta, out_dy);
}
